// TransformerBlock_45689862095177
// MI455X (gfx1250) — matrix operand order repaired and checked by reading, not compiled, not run on hardware
//
#include <hip/hip_runtime.h>

// ---------------------------------------------------------------------------
// Types for CDNA5 WMMA (gfx1250, wave32)
// ---------------------------------------------------------------------------
typedef __bf16 bf16;
typedef __attribute__((ext_vector_type(16))) __bf16 v16bf;
typedef __attribute__((ext_vector_type(8)))  __bf16 v8bf;
typedef __attribute__((ext_vector_type(8)))  float   v8f;

#define WMMA_BF16(a, b, c) \
  __builtin_amdgcn_wmma_f32_16x16x32_bf16(false, (a), false, (b), (short)0, (c), false, false)

static __device__ __forceinline__ v16bf concat8(v8bf lo, v8bf hi) {
  v16bf r;
#pragma unroll
  for (int i = 0; i < 8; ++i) { r[i] = lo[i]; r[i + 8] = hi[i]; }
  return r;
}

// ---------------------------------------------------------------------------
// fp32 -> bf16 cast
// ---------------------------------------------------------------------------
__global__ void cast_f32_to_bf16(const float* __restrict__ in, bf16* __restrict__ out, int n) {
  int i = blockIdx.x * 256 + threadIdx.x;
  if (i < n) out[i] = (bf16)in[i];
}

// ---------------------------------------------------------------------------
// LayerNorm (f32 in, bf16 out), one block per row, D = 1024
// ---------------------------------------------------------------------------
__global__ __launch_bounds__(256) void ln_f32_to_bf16(
    const float* __restrict__ x, const float* __restrict__ g, const float* __restrict__ b,
    bf16* __restrict__ out, int D) {
  __shared__ float red[256];
  const int tid = threadIdx.x;
  const float* xr = x + (size_t)blockIdx.x * D;

  float s = 0.f;
  for (int i = tid; i < D; i += 256) s += xr[i];
  red[tid] = s; __syncthreads();
  for (int off = 128; off; off >>= 1) { if (tid < off) red[tid] += red[tid + off]; __syncthreads(); }
  const float mean = red[0] / (float)D;
  __syncthreads();

  float v = 0.f;
  for (int i = tid; i < D; i += 256) { float d = xr[i] - mean; v += d * d; }
  red[tid] = v; __syncthreads();
  for (int off = 128; off; off >>= 1) { if (tid < off) red[tid] += red[tid + off]; __syncthreads(); }
  const float rstd = rsqrtf(red[0] / (float)D + 1e-5f);

  bf16* orow = out + (size_t)blockIdx.x * D;
  for (int i = tid; i < D; i += 256)
    orow[i] = (bf16)((xr[i] - mean) * rstd * g[i] + b[i]);
}

// ---------------------------------------------------------------------------
// bf16 GEMM with WMMA: C[M,N] = A[M,K] @ W[K,N] (+bias)(+GELU)(+residual)
// Block = 256 thr = 8 waves; block tile 128x128; wave tile 64x32 (4x2 WMMA).
// A tile LDS row-major [128][32]; B tile LDS transposed [n:128][k:32] so both
// fragment loads are 2x ds_load_b128 matching the ISA 16-bit A/B layouts.
// FLAGS: 1=bias, 2=gelu(exact), 4=residual, 8=bf16 out (else f32 out)
// MG: row bounds checks (only needed when M % 128 != 0)
// ---------------------------------------------------------------------------
template <int FLAGS, bool MG>
__global__ __launch_bounds__(256) void gemm_bf16_wmma(
    const bf16* __restrict__ A, const bf16* __restrict__ W,
    const float* __restrict__ bias, const float* __restrict__ res,
    void* __restrict__ outp, int M, int N, int K) {
  __shared__ __align__(16) bf16 As[128][32];
  __shared__ __align__(16) bf16 Bs[128][32];  // [n][k]

  const int tid  = threadIdx.x;
  const int m0   = blockIdx.y * 128;
  const int n0   = blockIdx.x * 128;
  const int wave = tid >> 5;
  const int lane = tid & 31;
  const int wm   = (wave >> 2) * 64;
  const int wn   = (wave & 3) * 32;
  const int lh   = lane >> 4;   // lane-half selects K group
  const int ln16 = lane & 15;   // M/N index within fragment

  const int arow = tid >> 1;               // A: 2 threads per 32-bf16 row
  const int acol = (tid & 1) * 16;
  const int brow = tid >> 3;               // B: thread -> W row (k), 32 rows
  const int bcol = (tid & 7) * 16;         //    16-wide N slice

  v8f acc[4][2];
#pragma unroll
  for (int mi = 0; mi < 4; ++mi)
#pragma unroll
    for (int ni = 0; ni < 2; ++ni) acc[mi][ni] = (v8f){0.f,0.f,0.f,0.f,0.f,0.f,0.f,0.f};

  const bf16* aptr = A + (size_t)(m0 + arow) * K + acol;
  const bf16* wptr = W + (size_t)brow * N + n0 + bcol;
  const bool  arow_ok = !MG || (m0 + arow < M);

  for (int k0 = 0; k0 < K; k0 += 32) {
    // ---- stage A tile: coalesced b128 copies ----
    {
      v8bf a0 = (v8bf){};
      v8bf a1 = (v8bf){};
      if (arow_ok) {
        a0 = *(const v8bf*)(aptr + k0);
        a1 = *(const v8bf*)(aptr + k0 + 8);
      }
      *(v8bf*)&As[arow][acol]     = a0;
      *(v8bf*)&As[arow][acol + 8] = a1;
    }
    // ---- stage B tile: coalesced b128 W reads + LDS scatter transpose ----
    {
      const bf16* wp = wptr + (size_t)k0 * N;
      v8bf w0 = *(const v8bf*)wp;
      v8bf w1 = *(const v8bf*)(wp + 8);
#pragma unroll
      for (int i = 0; i < 8; ++i) {
        Bs[bcol + i][brow]     = w0[i];
        Bs[bcol + 8 + i][brow] = w1[i];
      }
    }
    __syncthreads();

    // ---- fragments: 2x ds_load_b128 each ----
    v16bf afr[4];
#pragma unroll
    for (int mi = 0; mi < 4; ++mi) {
      int row = wm + mi * 16 + ln16;
      int kb  = lh * 8;
      v8bf lo = *(const v8bf*)&As[row][kb];
      v8bf hi = *(const v8bf*)&As[row][16 + kb];
      afr[mi] = concat8(lo, hi);
    }
    v16bf bfr[2];
#pragma unroll
    for (int ni = 0; ni < 2; ++ni) {
      int n   = wn + ni * 16 + ln16;
      int kb2 = lh * 8;
      v8bf lo = *(const v8bf*)&Bs[n][kb2];
      v8bf hi = *(const v8bf*)&Bs[n][16 + kb2];
      bfr[ni] = concat8(lo, hi);
    }
#pragma unroll
    for (int mi = 0; mi < 4; ++mi)
#pragma unroll
      for (int ni = 0; ni < 2; ++ni)
        acc[mi][ni] = WMMA_BF16(afr[mi], bfr[ni], acc[mi][ni]);
    __syncthreads();
  }

  // ---- fused epilogue ----
#pragma unroll
  for (int mi = 0; mi < 4; ++mi)
#pragma unroll
    for (int ni = 0; ni < 2; ++ni)
#pragma unroll
      for (int r = 0; r < 8; ++r) {
        int row = m0 + wm + mi * 16 + lh * 8 + r;
        int col = n0 + wn + ni * 16 + ln16;
        if (!MG || row < M) {
          float v = acc[mi][ni][r];
          if (FLAGS & 1) v += bias[col];
          if (FLAGS & 2) v = 0.5f * v * (1.f + erff(v * 0.70710678118654752f));
          if (FLAGS & 4) v += res[(size_t)row * N + col];
          if (FLAGS & 8) ((bf16*)outp)[(size_t)row * N + col] = (bf16)v;
          else           ((float*)outp)[(size_t)row * N + col] = v;
        }
      }
}

// ---------------------------------------------------------------------------
// Flash attention (bf16 in/out, f32 accum) — one wave per 32 q-rows per (b,h).
// DH = 64. Per 32-key tile: 8 WMMA (S = Q.K^T), 8 WMMA (P.V), 2 WMMA
// (row-sums via P @ ones). K/V fragments are reused by both 16-row M-frags.
// Online softmax keeps only the row-max cross-lane reduction (batched so each
// round needs a single dscnt wait); row sums come out of the ones-WMMA in
// C-fragment layout, pre-scaled by the same alpha chain as O.
// OOB keys: index clamped with min(); invalid columns masked to -1e30 so P=0.
// ---------------------------------------------------------------------------
__global__ __launch_bounds__(32) void attn_wmma(
    const bf16* __restrict__ Q, int qrs, int qc0,
    const bf16* __restrict__ Kp, int krs, int kc0,
    const bf16* __restrict__ Vp, int vrs, int vc0,
    bf16* __restrict__ O, int ors, int oc0,
    int Tq, int Tk, int causal) {
  __shared__ __align__(16) bf16 Ps[32][32];

  const int lane = threadIdx.x;
  const int qt = blockIdx.x, h = blockIdx.y, b = blockIdx.z;
  const int ln16 = lane & 15, lh = lane >> 4;

  const bf16* __restrict__ Qb = Q  + (size_t)b * Tq * qrs + (qc0 + h * 64);
  const bf16* __restrict__ Kb = Kp + (size_t)b * Tk * krs + (kc0 + h * 64);
  const bf16* __restrict__ Vb = Vp + (size_t)b * Tk * vrs + (vc0 + h * 64);
  bf16* __restrict__       Ob = O  + (size_t)b * Tq * ors + (oc0 + h * 64);

  // Q A-fragments: 2 M-frags x 2 dh-chunks, contiguous b128 loads
  v16bf qa[2][2];
#pragma unroll
  for (int mi = 0; mi < 2; ++mi) {
    const int qoff = (qt * 32 + mi * 16 + ln16) * qrs + lh * 8;
#pragma unroll
    for (int f = 0; f < 2; ++f) {
      v8bf lo = *(const v8bf*)(Qb + qoff + f * 32);
      v8bf hi = *(const v8bf*)(Qb + qoff + f * 32 + 16);
      qa[mi][f] = concat8(lo, hi);
    }
  }

  // all-ones B-fragment for row-sum WMMA
  v16bf ones;
#pragma unroll
  for (int e = 0; e < 16; ++e) ones[e] = (bf16)1.0f;

  float mrow[2][8];
  v8f o[2][4], lacc[2];
#pragma unroll
  for (int mi = 0; mi < 2; ++mi) {
#pragma unroll
    for (int r = 0; r < 8; ++r) mrow[mi][r] = -1e30f;
#pragma unroll
    for (int d = 0; d < 4; ++d) o[mi][d] = (v8f){0.f,0.f,0.f,0.f,0.f,0.f,0.f,0.f};
    lacc[mi] = (v8f){0.f,0.f,0.f,0.f,0.f,0.f,0.f,0.f};
  }

  const int sEnd = causal ? (qt * 32 + 32) : Tk;
  for (int sb = 0; sb < sEnd; sb += 32) {
    // ---- K^T B-fragments (shared by both M-frags) ----
    v16bf kf[2][2];
#pragma unroll
    for (int t = 0; t < 2; ++t) {
      int s  = min(sb + t * 16 + ln16, Tk - 1);
      int ko = s * krs + lh * 8;
#pragma unroll
      for (int d = 0; d < 2; ++d) {
        v8bf lo = *(const v8bf*)(Kb + ko + d * 32);
        v8bf hi = *(const v8bf*)(Kb + ko + d * 32 + 16);
        kf[t][d] = concat8(lo, hi);
      }
    }
    const v8f zero = (v8f){0.f,0.f,0.f,0.f,0.f,0.f,0.f,0.f};
    v8f sf[2][2];
#pragma unroll
    for (int mi = 0; mi < 2; ++mi)
#pragma unroll
      for (int t = 0; t < 2; ++t) {
        sf[mi][t] = WMMA_BF16(qa[mi][0], kf[t][0], zero);
        sf[mi][t] = WMMA_BF16(qa[mi][1], kf[t][1], sf[mi][t]);
      }

    // ---- scale + mask; tile row-max (batched cross-lane reduction) ----
    const int c0 = sb + ln16, c1 = sb + 16 + ln16;
    float sv0[2][8], sv1[2][8], mt[2][8];
#pragma unroll
    for (int mi = 0; mi < 2; ++mi)
#pragma unroll
      for (int r = 0; r < 8; ++r) {
        int qg = qt * 32 + mi * 16 + lh * 8 + r;
        float v0 = sf[mi][0][r] * 0.125f;
        float v1 = sf[mi][1][r] * 0.125f;
        bool ok0 = (c0 < Tk) && (!causal || c0 <= qg);
        bool ok1 = (c1 < Tk) && (!causal || c1 <= qg);
        sv0[mi][r] = ok0 ? v0 : -1e30f;
        sv1[mi][r] = ok1 ? v1 : -1e30f;
        mt[mi][r] = fmaxf(sv0[mi][r], sv1[mi][r]);
      }
#pragma unroll
    for (int off = 1; off < 16; off <<= 1) {
      float tmp[2][8];
#pragma unroll
      for (int mi = 0; mi < 2; ++mi)
#pragma unroll
        for (int r = 0; r < 8; ++r) tmp[mi][r] = __shfl_xor(mt[mi][r], off);
#pragma unroll
      for (int mi = 0; mi < 2; ++mi)
#pragma unroll
        for (int r = 0; r < 8; ++r) mt[mi][r] = fmaxf(mt[mi][r], tmp[mi][r]);
    }

    // ---- online update: rescale O & lacc, exponentiate, stage P ----
#pragma unroll
    for (int mi = 0; mi < 2; ++mi)
#pragma unroll
      for (int r = 0; r < 8; ++r) {
        float mnew  = fmaxf(mrow[mi][r], mt[mi][r]);
        float alpha = __expf(mrow[mi][r] - mnew);
        mrow[mi][r] = mnew;
        float p0 = __expf(sv0[mi][r] - mnew);
        float p1 = __expf(sv1[mi][r] - mnew);
#pragma unroll
        for (int d = 0; d < 4; ++d) o[mi][d][r] *= alpha;
        lacc[mi][r] *= alpha;
        Ps[mi * 16 + lh * 8 + r][ln16]      = (bf16)p0;
        Ps[mi * 16 + lh * 8 + r][16 + ln16] = (bf16)p1;
      }
    __syncthreads();

    // ---- P: C-frag -> A-frag via LDS (2x ds_load_b128 per M-frag) ----
    v16bf pa[2];
#pragma unroll
    for (int mi = 0; mi < 2; ++mi) {
      int kb = lh * 16;
      v8bf lo = *(const v8bf*)&Ps[mi * 16 + ln16][kb];
      v8bf hi = *(const v8bf*)&Ps[mi * 16 + ln16][kb + 8];
      pa[mi] = concat8(lo, hi);
    }

    // ---- V B-fragments (shared): clamped row offsets computed once ----
    int voff[16];
#pragma unroll
    for (int e = 0; e < 16; ++e)
      voff[e] = min(sb + lh * 16 + e, Tk - 1) * vrs;
#pragma unroll
    for (int dn = 0; dn < 4; ++dn) {
      const int col = dn * 16 + ln16;
      v16bf vb;
#pragma unroll
      for (int e = 0; e < 16; ++e) vb[e] = Vb[voff[e] + col];
#pragma unroll
      for (int mi = 0; mi < 2; ++mi)
        o[mi][dn] = WMMA_BF16(pa[mi], vb, o[mi][dn]);
    }
    // row-sums: lacc += P @ ones (every column of lacc = rowsum)
#pragma unroll
    for (int mi = 0; mi < 2; ++mi)
      lacc[mi] = WMMA_BF16(pa[mi], ones, lacc[mi]);
    __syncthreads();
  }

  // ---- write O / l ----
#pragma unroll
  for (int mi = 0; mi < 2; ++mi)
#pragma unroll
    for (int dn = 0; dn < 4; ++dn)
#pragma unroll
      for (int r = 0; r < 8; ++r) {
        int row = qt * 32 + mi * 16 + lh * 8 + r;
        Ob[(size_t)row * ors + dn * 16 + ln16] = (bf16)(o[mi][dn][r] / lacc[mi][r]);
      }
}

// ---------------------------------------------------------------------------
// Orchestration
// ---------------------------------------------------------------------------
extern "C" void kernel_launch(void* const* d_in, const int* in_sizes, int n_in,
                              void* d_out, int out_size, void* d_ws, size_t ws_size,
                              hipStream_t stream) {
  constexpr int B = 2, T = 2048, D = 1024, H = 16, DFF = 4096, DC = 768, C = 77;
  constexpr int NT = B * T;        // 4096 token rows
  constexpr int NC = B * C;        // 154 cond rows

  const float* x        = (const float*)d_in[0];
  const float* cond     = (const float*)d_in[1];
  const float* Wqkv     = (const float*)d_in[2];
  const float* Wproj_sa = (const float*)d_in[3];
  const float* bproj_sa = (const float*)d_in[4];
  const float* g1       = (const float*)d_in[5];
  const float* b1       = (const float*)d_in[6];
  const float* Wq_ca    = (const float*)d_in[7];
  const float* Wkv_ca   = (const float*)d_in[8];
  const float* Wproj_ca = (const float*)d_in[9];
  const float* bproj_ca = (const float*)d_in[10];
  const float* g2       = (const float*)d_in[11];
  const float* b2       = (const float*)d_in[12];
  const float* Wff1     = (const float*)d_in[13];
  const float* bff1     = (const float*)d_in[14];
  const float* Wff2     = (const float*)d_in[15];
  const float* bff2     = (const float*)d_in[16];
  const float* g3       = (const float*)d_in[17];
  const float* b3       = (const float*)d_in[18];

  char* base = (char*)d_ws;
  size_t off = 0;
  auto alloc = [&](size_t bytes) -> void* {
    void* p = base + off;
    off += (bytes + 255) & ~(size_t)255;
    return p;
  };

  bf16* wqkv_b  = (bf16*)alloc((size_t)D * 3 * D * 2);
  bf16* wpsa_b  = (bf16*)alloc((size_t)D * D * 2);
  bf16* wqca_b  = (bf16*)alloc((size_t)D * D * 2);
  bf16* wkvca_b = (bf16*)alloc((size_t)DC * 2 * D * 2);
  bf16* wpca_b  = (bf16*)alloc((size_t)D * D * 2);
  bf16* wff1_b  = (bf16*)alloc((size_t)D * DFF * 2);
  bf16* wff2_b  = (bf16*)alloc((size_t)DFF * D * 2);
  bf16* h_b     = (bf16*)alloc((size_t)NT * D * 2);      // layernorm output (reused)
  bf16* qkv_b   = (bf16*)alloc((size_t)NT * 3 * D * 2);  // self-attn qkv
  bf16* attn_b  = (bf16*)alloc((size_t)NT * D * 2);      // attention output (reused)
  bf16* q_b     = (bf16*)alloc((size_t)NT * D * 2);      // cross-attn q
  bf16* cond_b  = (bf16*)alloc((size_t)NC * DC * 2);
  bf16* kv_b    = (bf16*)alloc((size_t)NC * 2 * D * 2);
  bf16* ffh_b   = (bf16*)alloc((size_t)NT * DFF * 2);
  float* x1     = (float*)alloc((size_t)NT * D * 4);
  float* x2     = (float*)alloc((size_t)NT * D * 4);
  (void)ws_size; (void)n_in; (void)in_sizes; (void)out_size;

  auto cast = [&](const float* s, bf16* d, int n) {
    cast_f32_to_bf16<<<(n + 255) / 256, 256, 0, stream>>>(s, d, n);
  };
  cast(Wqkv,     wqkv_b,  D * 3 * D);
  cast(Wproj_sa, wpsa_b,  D * D);
  cast(Wq_ca,    wqca_b,  D * D);
  cast(Wkv_ca,   wkvca_b, DC * 2 * D);
  cast(Wproj_ca, wpca_b,  D * D);
  cast(Wff1,     wff1_b,  D * DFF);
  cast(Wff2,     wff2_b,  DFF * D);
  cast(cond,     cond_b,  NC * DC);

  dim3 blk(256);
#define GEMM(FL, MG, A_, W_, BIAS_, RES_, OUT_, M_, N_, K_)                           \
  gemm_bf16_wmma<FL, MG><<<dim3((N_) / 128, ((M_) + 127) / 128), blk, 0, stream>>>(  \
      A_, W_, BIAS_, RES_, OUT_, M_, N_, K_)

  // ---- self-attention ----
  ln_f32_to_bf16<<<NT, 256, 0, stream>>>(x, g1, b1, h_b, D);
  GEMM(8, false, h_b, wqkv_b, nullptr, nullptr, qkv_b, NT, 3 * D, D);     // qkv (bf16)
  attn_wmma<<<dim3(T / 32, H, B), 32, 0, stream>>>(                        // causal SA
      qkv_b, 3 * D, 0, qkv_b, 3 * D, D, qkv_b, 3 * D, 2 * D,
      attn_b, D, 0, T, T, 1);
  GEMM(1 | 4, false, attn_b, wpsa_b, bproj_sa, x, x1, NT, D, D);           // proj + res -> x1

  // ---- cross-attention ----
  ln_f32_to_bf16<<<NT, 256, 0, stream>>>(x1, g2, b2, h_b, D);
  GEMM(8, false, h_b, wqca_b, nullptr, nullptr, q_b, NT, D, D);            // q (bf16)
  GEMM(8, true, cond_b, wkvca_b, nullptr, nullptr, kv_b, NC, 2 * D, DC);   // kv (bf16, M=154)
  attn_wmma<<<dim3(T / 32, H, B), 32, 0, stream>>>(                        // non-causal CA
      q_b, D, 0, kv_b, 2 * D, 0, kv_b, 2 * D, D,
      attn_b, D, 0, T, C, 0);
  GEMM(1 | 4, false, attn_b, wpca_b, bproj_ca, x1, x2, NT, D, D);          // proj + res -> x2

  // ---- feed-forward ----
  ln_f32_to_bf16<<<NT, 256, 0, stream>>>(x2, g3, b3, h_b, D);
  GEMM(1 | 2 | 8, false, h_b, wff1_b, bff1, nullptr, ffh_b, NT, DFF, D);   // ff1 + GELU (bf16)
  GEMM(1 | 4, false, ffh_b, wff2_b, bff2, x2, (float*)d_out, NT, D, DFF);  // ff2 + res -> out
#undef GEMM
}
